// Attention_17532056502764
// MI455X (gfx1250) — matrix operand order repaired and checked by reading, not compiled, not run on hardware
//
#include <hip/hip_runtime.h>
#include <hip/hip_bf16.h>

// ---------------- problem constants ----------------
#define BSZ      2
#define SEQ      2048
#define DIMI     1024
#define HEADS    16
#define DH       64
#define NM       16
#define NKV      2064          // NM + SEQ
#define NKV_PAD  2112          // 33 * 64 (padded for j-tiling)
#define JTILE    64
#define NEG_MAX  (-3.402823466e+38f)
#define SCALE_QK 0.125f        // 64^-0.5

// ---------------- CDNA5 WMMA types ----------------
typedef __attribute__((ext_vector_type(16))) __bf16 v16bf;
typedef __attribute__((ext_vector_type(8)))  float  v8f;
typedef __attribute__((ext_vector_type(2)))  float  v2f;

struct U256 { uint4 lo, hi; };

static __device__ inline v16bf make_frag(uint4 a, uint4 b) {
    U256 u{a, b};
    return __builtin_bit_cast(v16bf, u);
}

// f32 -> bf16 bits, round-to-nearest-even (storage as ushort; no scalar __bf16 math)
static __device__ inline unsigned short f2bf(float f) {
    unsigned int u = __builtin_bit_cast(unsigned int, f);
    u = (u + 0x7FFFu + ((u >> 16) & 1u)) >> 16;
    return (unsigned short)u;
}

// CDNA5 async copy: 16B global -> LDS, tracked by ASYNCcnt (gfx1250).
// ldsaddr = low 32 bits of a generic pointer into LDS (LDS aperture keeps the
// byte offset in addr[31:0] per CDNA5 ISA 10.2); gaddr = 64-bit global address.
static __device__ inline void async_g2l_b128(unsigned ldsaddr, unsigned long long gaddr) {
    asm volatile("global_load_async_to_lds_b128 %0, %1, off"
                 :: "v"(ldsaddr), "v"(gaddr) : "memory");
}
static __device__ inline void wait_asynccnt0() {
    asm volatile("s_wait_asynccnt 0" ::: "memory");
}

// ======================================================================
// Fill memory-KV slots (rows 0..NM-1) of Kbf (bf16) and Vf (f32), both b.
// ======================================================================
__global__ void memfill_kernel(const float* __restrict__ mem_k,
                               const float* __restrict__ mem_v,
                               unsigned short* __restrict__ Kbf,
                               float* __restrict__ Vf) {
    int idx = blockIdx.x * 256 + threadIdx.x;       // BSZ*HEADS*NM*DH = 32768
    if (idx >= BSZ * HEADS * NM * DH) return;
    int d = idx & 63;
    int s = (idx >> 6) & 15;
    int h = (idx >> 10) & 15;
    int b = idx >> 14;
    size_t src = ((size_t)h * NM + s) * DH + d;
    size_t dst = (((size_t)(b * HEADS + h)) * NKV_PAD + s) * DH + d;
    Kbf[dst] = f2bf(mem_k[src]);
    Vf[dst]  = mem_v[src];
}

// ======================================================================
// WMMA GEMM: C[64x64 tile] = X(MxK) * W(KxN), K = N = 1024, M = 4096.
// MODE 0: X=f32, out = Qbf (bf16, [b][h][i][d])
// MODE 1: X=f32, out = Kbf (bf16, [b][h][NM+i][d])
// MODE 2: X=f32, out = Vf  (f32,  [b][h][NM+i][d])
// MODE 3: X=bf16 (AObf, async-DMA staged), out = d_out f32 with bias add
// Block = 128 threads (4 waves); each wave owns a 16x64 strip.
// ======================================================================
template <int MODE>
__global__ __launch_bounds__(128)
void gemm64_kernel(const void* __restrict__ Xv, const float* __restrict__ W,
                   void* __restrict__ Out, const float* __restrict__ bias) {
    __shared__ __align__(16) unsigned short Abf[64][40];   // [m][k] bf16 tile
    __shared__ __align__(16) unsigned short Btr[64][40];   // [n][k] bf16 tile (transposed)

    const int t    = threadIdx.x;
    const int wave = t >> 5;
    const int lane = t & 31;
    const int half = lane >> 4;
    const int lid  = lane & 15;
    const int mBase = blockIdx.x * 64;
    const int nBase = blockIdx.y * 64;

    v8f acc[4] = {};

    for (int k0 = 0; k0 < DIMI; k0 += 32) {
        // ---- stage X tile (64 x 32) into LDS as bf16 ----
        {
            int row = t >> 1;
            int c   = (t & 1) * 16;
            if (MODE == 3) {
                // already bf16: pure copy -> CDNA5 async global->LDS DMA (2x16B)
                const unsigned short* X = (const unsigned short*)Xv;
                const unsigned short* src = X + (size_t)(mBase + row) * DIMI + k0 + c;
                unsigned ldsdst = (unsigned)(size_t)(void*)&Abf[row][c];
                unsigned long long ga = (unsigned long long)(size_t)src;
                async_g2l_b128(ldsdst,      ga);
                async_g2l_b128(ldsdst + 16, ga + 16);
            } else {
                const float* X = (const float*)Xv;
                const float4* src = (const float4*)(X + (size_t)(mBase + row) * DIMI + k0 + c);
#pragma unroll
                for (int q = 0; q < 4; ++q) {
                    float4 f = src[q];
                    Abf[row][c + q * 4 + 0] = f2bf(f.x);
                    Abf[row][c + q * 4 + 1] = f2bf(f.y);
                    Abf[row][c + q * 4 + 2] = f2bf(f.z);
                    Abf[row][c + q * 4 + 3] = f2bf(f.w);
                }
            }
        }
        // ---- stage W tile (32 x 64) transposed into LDS (float4 loads) ----
#pragma unroll
        for (int it = 0; it < 4; ++it) {
            int idx = t + it * 128;            // 512 float4 = 32k x 64n
            int kk = idx >> 4;                 // 16 float4 per k-row
            int n4 = (idx & 15) * 4;
            float4 f = *(const float4*)&W[(size_t)(k0 + kk) * DIMI + nBase + n4];
            Btr[n4 + 0][kk] = f2bf(f.x);
            Btr[n4 + 1][kk] = f2bf(f.y);
            Btr[n4 + 2][kk] = f2bf(f.z);
            Btr[n4 + 3][kk] = f2bf(f.w);
        }
        if (MODE == 3) wait_asynccnt0();
        __syncthreads();

        // ---- A fragment for this wave's 16-row strip ----
        const int mRow = wave * 16 + lid;
        v16bf afrag;
        {
            uint4 a0 = *(const uint4*)&Abf[mRow][half * 8];
            uint4 a1 = *(const uint4*)&Abf[mRow][16 + half * 8];
            afrag = make_frag(a0, a1);
        }
#pragma unroll
        for (int s = 0; s < 4; ++s) {
            const unsigned short* bp = &Btr[s * 16 + lid][half * 8];
            v16bf bfrag = make_frag(*(const uint4*)bp, *(const uint4*)(bp + 16));
            acc[s] = __builtin_amdgcn_wmma_f32_16x16x32_bf16(
                false, afrag, false, bfrag, (short)0, acc[s], false, false);
        }
        __syncthreads();
    }

    // ---- write out in the destination layout ----
#pragma unroll
    for (int s = 0; s < 4; ++s) {
#pragma unroll
        for (int r = 0; r < 8; ++r) {
            int m  = r + half * 8;
            int gm = mBase + wave * 16 + m;
            int gc = nBase + s * 16 + lid;
            float v = acc[s][r];
            if (MODE == 3) {
                ((float*)Out)[(size_t)gm * DIMI + gc] = v + bias[gc];
            } else {
                int bb = gm >> 11, ii = gm & 2047;
                int hh = gc >> 6,  dd = gc & 63;
                if (MODE == 2) {
                    ((float*)Out)[(((size_t)(bb * HEADS + hh)) * NKV_PAD + NM + ii) * DH + dd] = v;
                } else if (MODE == 1) {
                    ((unsigned short*)Out)[(((size_t)(bb * HEADS + hh)) * NKV_PAD + NM + ii) * DH + dd] = f2bf(v);
                } else {
                    ((unsigned short*)Out)[(((size_t)(bb * HEADS + hh)) * SEQ + ii) * DH + dd] = f2bf(v);
                }
            }
        }
    }
}

// ======================================================================
// Fused attention: per (b, 16-row query tile).
//  A: per-head scores via bf16 WMMA (inner DH=64) -> LDS (f32, scaled)
//  B: pre_proj head-mix via f32 WMMA (16x16x4), causal/pad mask -> LDS
//  C: running top-8 per (k-head, row) in registers
//  D: softmax over the 8 survivors
//  E: post_proj head-mix + 8-sparse attn @ V, write bf16 [b][i][h*dh]
// Block = 256 threads (8 waves). Dynamic LDS = 144 KB.
// ======================================================================
__global__ __launch_bounds__(256)
void attn_kernel(const unsigned short* __restrict__ Qbf,
                 const unsigned short* __restrict__ Kbf,
                 const float* __restrict__ Vf,
                 const float* __restrict__ pre,
                 const float* __restrict__ post,
                 unsigned short* __restrict__ AObf) {
    extern __shared__ char smem[];
    float* s_lds = (float*)smem;                       // [16h][1024c]  c = i*64+jt
    float* m_lds = (float*)(smem + 65536);             // [16k][1024c]
    float* pval  = (float*)(smem + 131072);            // [16k][16i][8]
    int*   pidx  = (int*)(smem + 131072 + 8192);       // [16k][16i][8]

    const int t    = threadIdx.x;
    const int wave = t >> 5;
    const int lane = t & 31;
    const int half = lane >> 4;
    const int lid  = lane & 15;
    const int b     = blockIdx.x >> 7;                 // 128 i-tiles per batch
    const int iBase = (blockIdx.x & 127) * 16;

    // pre_proj^T fragments for f32 WMMA:  A[k][h] = pre[h][k]
    v2f pf[4];
#pragma unroll
    for (int ks = 0; ks < 4; ++ks) {
        int h0 = ks * 4 + half * 2;                    // f32 A: k(inner) = r + 2*half
        pf[ks].x = pre[(h0 + 0) * HEADS + lid];
        pf[ks].y = pre[(h0 + 1) * HEADS + lid];
    }

    // Q fragments: wave handles heads {2*wave, 2*wave+1}; A 16x32 bf16 x2 chunks
    v16bf qf[2][2];
#pragma unroll
    for (int hh = 0; hh < 2; ++hh) {
        int h = wave * 2 + hh;
        const unsigned short* qrow =
            Qbf + (((size_t)b * HEADS + h) * SEQ + iBase + lid) * DH;
#pragma unroll
        for (int c = 0; c < 2; ++c) {
            const unsigned short* p0 = qrow + c * 32 + half * 8;
            const unsigned short* p1 = qrow + c * 32 + 16 + half * 8;
            qf[hh][c] = make_frag(*(const uint4*)p0, *(const uint4*)p1);
        }
    }

    // per-thread top-8 state for row (kq, iq)
    const int kq = t >> 4, iq = t & 15;
    float tv[8]; int tj[8];
#pragma unroll
    for (int e = 0; e < 8; ++e) { tv[e] = NEG_MAX; tj[e] = 0; }

    for (int jBase = 0; jBase < NKV_PAD; jBase += JTILE) {
        // ---------- Stage A: per-head scores ----------
#pragma unroll
        for (int hh = 0; hh < 2; ++hh) {
            int h = wave * 2 + hh;
            const unsigned short* kbase = Kbf + ((size_t)b * HEADS + h) * NKV_PAD * DH;
#pragma unroll
            for (int js = 0; js < 4; ++js) {
                int jpos = jBase + js * 16 + lid;
                const unsigned short* krow = kbase + (size_t)jpos * DH;
                v8f acc = {};
#pragma unroll
                for (int c = 0; c < 2; ++c) {          // inner d = c*32 + half*8 + e (e<8), c*32 + 16 + half*8 + (e-8)
                    const unsigned short* p0 = krow + c * 32 + half * 8;
                    v16bf kf = make_frag(*(const uint4*)p0, *(const uint4*)(p0 + 16));
                    acc = __builtin_amdgcn_wmma_f32_16x16x32_bf16(
                        false, qf[hh][c], false, kf, (short)0, acc, false, false);
                }
#pragma unroll
                for (int r = 0; r < 8; ++r) {
                    int i = r + half * 8;
                    s_lds[h * 1024 + i * 64 + js * 16 + lid] = acc[r] * SCALE_QK;
                }
            }
        }
        __syncthreads();

        // ---------- Stage B: pre_proj mix (f32 WMMA) + mask ----------
#pragma unroll
        for (int sub = 0; sub < 8; ++sub) {
            int cBase = (wave * 8 + sub) * 16;
            v8f acc = {};
#pragma unroll
            for (int ks = 0; ks < 4; ++ks) {
                int h0 = ks * 4 + half * 2;            // f32 B: k(inner) = r + 2*half
                v2f bfr;
                bfr.x = s_lds[(h0 + 0) * 1024 + cBase + lid];
                bfr.y = s_lds[(h0 + 1) * 1024 + cBase + lid];
                acc = __builtin_amdgcn_wmma_f32_16x16x4_f32(
                    false, pf[ks], false, bfr, (short)0, acc, false, false);
            }
            int col  = cBase + lid;
            int i    = col >> 6;
            int jt   = col & 63;
            int jpos = jBase + jt;
            bool masked = (jpos >= NKV) || (jpos > iBase + i + NM);   // causal + pad
#pragma unroll
            for (int r = 0; r < 8; ++r) {
                int k = r + half * 8;
                m_lds[k * 1024 + col] = masked ? NEG_MAX : acc[r];
            }
        }
        __syncthreads();

        // ---------- Stage C: running top-8 ----------
        {
            const float* row = &m_lds[kq * 1024 + iq * 64];
            for (int jt = 0; jt < JTILE; ++jt) {
                float v = row[jt];
                if (v > tv[7]) {
                    tv[7] = v; tj[7] = jBase + jt;
#pragma unroll
                    for (int e = 7; e > 0; --e) {
                        bool sw = tv[e] > tv[e - 1];
                        float fv = sw ? tv[e - 1] : tv[e];
                        tv[e - 1] = sw ? tv[e] : tv[e - 1];
                        tv[e] = fv;
                        int fj = sw ? tj[e - 1] : tj[e];
                        tj[e - 1] = sw ? tj[e] : tj[e - 1];
                        tj[e] = fj;
                    }
                }
            }
        }
        __syncthreads();
    }

    // ---------- Stage D: softmax over the 8 kept entries ----------
    {
        float m = tv[0];
        float p[8];
        float z = 0.f;
#pragma unroll
        for (int e = 0; e < 8; ++e) { p[e] = __expf(tv[e] - m); z += p[e]; }
        float rz = 1.0f / z;
#pragma unroll
        for (int e = 0; e < 8; ++e) {
            pval[(kq * 16 + iq) * 8 + e] = p[e] * rz;
            pidx[(kq * 16 + iq) * 8 + e] = tj[e];
        }
    }
    __syncthreads();

    // ---------- Stage E: post mix + sparse attn @ V ----------
    {
        const int k2 = kq, i = iq;
        float4 out4[16];
#pragma unroll
        for (int q = 0; q < 16; ++q) out4[q] = make_float4(0.f, 0.f, 0.f, 0.f);
        const float* vbase = Vf + ((size_t)b * HEADS + k2) * NKV_PAD * DH;
        for (int k = 0; k < HEADS; ++k) {
            float wp = post[k * HEADS + k2];
#pragma unroll
            for (int e = 0; e < 8; ++e) {
                float pw = pval[(k * 16 + i) * 8 + e] * wp;
                int   j  = pidx[(k * 16 + i) * 8 + e];
                const float4* vrow = (const float4*)(vbase + (size_t)j * DH);
#pragma unroll
                for (int q = 0; q < 16; ++q) {
                    float4 vv = vrow[q];
                    out4[q].x += pw * vv.x;
                    out4[q].y += pw * vv.y;
                    out4[q].z += pw * vv.z;
                    out4[q].w += pw * vv.w;
                }
            }
        }
        unsigned short* dst = AObf + ((size_t)b * SEQ + iBase + i) * DIMI + k2 * DH;
#pragma unroll
        for (int q = 0; q < 16; ++q) {
            dst[q * 4 + 0] = f2bf(out4[q].x);
            dst[q * 4 + 1] = f2bf(out4[q].y);
            dst[q * 4 + 2] = f2bf(out4[q].z);
            dst[q * 4 + 3] = f2bf(out4[q].w);
        }
    }
}

// ======================================================================
extern "C" void kernel_launch(void* const* d_in, const int* in_sizes, int n_in,
                              void* d_out, int out_size, void* d_ws, size_t ws_size,
                              hipStream_t stream) {
    (void)in_sizes; (void)n_in; (void)out_size; (void)ws_size;
    const float* x     = (const float*)d_in[0];
    const float* Wq    = (const float*)d_in[1];
    const float* Wk    = (const float*)d_in[2];
    const float* Wv    = (const float*)d_in[3];
    const float* pre   = (const float*)d_in[4];
    const float* post  = (const float*)d_in[5];
    const float* mem_k = (const float*)d_in[6];
    const float* mem_v = (const float*)d_in[7];
    const float* Wo    = (const float*)d_in[8];
    const float* bo    = (const float*)d_in[9];

    char* ws = (char*)d_ws;
    const size_t szQ  = (size_t)BSZ * HEADS * SEQ * DH * 2;          //  8,388,608
    const size_t szK  = (size_t)BSZ * HEADS * NKV_PAD * DH * 2;      //  8,650,752
    const size_t szV  = (size_t)BSZ * HEADS * NKV_PAD * DH * 4;      // 17,301,504
    unsigned short* Qbf  = (unsigned short*)ws;
    unsigned short* Kbf  = (unsigned short*)(ws + szQ);
    float*          Vf   = (float*)(ws + szQ + szK);
    unsigned short* AObf = (unsigned short*)(ws + szQ + szK + szV);  // +8,388,608

    dim3 gGemm(64, 16), bGemm(128);

    memfill_kernel<<<128, 256, 0, stream>>>(mem_k, mem_v, Kbf, Vf);
    gemm64_kernel<0><<<gGemm, bGemm, 0, stream>>>(x, Wq, Qbf, nullptr);
    gemm64_kernel<1><<<gGemm, bGemm, 0, stream>>>(x, Wk, Kbf, nullptr);
    gemm64_kernel<2><<<gGemm, bGemm, 0, stream>>>(x, Wv, Vf, nullptr);

    // 144 KB dynamic LDS: scores(64K) + mixed(64K) + pval(8K) + pidx(8K)
    attn_kernel<<<dim3(BSZ * (SEQ / 16)), dim3(256), 147456, stream>>>(
        Qbf, Kbf, Vf, pre, post, AObf);

    gemm64_kernel<3><<<gGemm, bGemm, 0, stream>>>(AObf, Wo, d_out, bo);
}
